// IaBNet_with_affinity_62088047231277
// MI455X (gfx1250) — matrix operand order repaired and checked by reading, not compiled, not run on hardware
//
#include <hip/hip_runtime.h>
#include <hip/hip_bf16.h>

// ---------------------------------------------------------------------------
// MI455X (gfx1250). bf16 WMMA (v_wmma_f32_16x16x32_bf16) for all three GEMMs;
// fused softmax + distance-bucket bias + scrambled-mask epilogue. Intermediates
// bf16 in workspace (~42MB) so they stay L2-resident (192MB L2). The whole
// pipeline is bounded by the single 268MB fp32 read of z (~11.5us @ 23.3TB/s).
// All LDS operand tiles are stored pre-swizzled so every WMMA B-fragment is
// two ds_load_b128 (32 contiguous bytes per lane), never scalar u16 gathers.
// ---------------------------------------------------------------------------

typedef __attribute__((ext_vector_type(16))) __bf16 v16bf;
typedef __attribute__((ext_vector_type(8)))  float  v8f;

#define Bn 4
#define NP 512
#define NC 256
#define Ee 128
#define Cc 32

// Native f32->bf16 cvt path (lowered to v_cvt_pk_bf16_f32 pairs).
static __device__ __forceinline__ unsigned short f2bfbits(float f) {
    return __builtin_bit_cast(unsigned short, (__bf16)f);
}
static __device__ __forceinline__ float bf2f(unsigned short h) {
    unsigned u = ((unsigned)h) << 16;
    return __builtin_bit_cast(float, u);
}

// ---------------------------------------------------------------------------
// zero the 8 per-batch energy accumulators (graph replays must not accumulate)
// ---------------------------------------------------------------------------
__global__ void k_zero(float* acc) {
    if (threadIdx.x < 8) acc[threadIdx.x] = 0.0f;
}

// ---------------------------------------------------------------------------
// GEMM1: S[b,nc,c,np] = z[b,np,nc,:] @ W_lin[:,c] + b_lin[c]   (bf16 out)
// One block per (b,nc). W_lin staged into LDS pre-swizzled
// (ldsW[(n*8 + (k>>4))*16 + (k&15)]) so each of the 8 register-resident
// B-fragments is built with 2 ds_load_b128; fragments are then reused across
// all 4 np-strips of 128. Wave = 16(np) x 32(c) per strip.
// ---------------------------------------------------------------------------
__global__ void k_gemm1(const float* __restrict__ z, const float* __restrict__ Wl,
                        const float* __restrict__ bl, unsigned short* __restrict__ S) {
    __shared__ unsigned short ldsW[Ee * Cc];
    const int tid = threadIdx.x;
    {
        const int kTh = (tid * 16) >> 5;     // fixed k for this thread's 16 elems
        const int nTh = (tid * 16) & 31;     // 16 consecutive n
#pragma unroll
        for (int i = 0; i < 16; ++i) {
            const int n = nTh + i;
            ldsW[(n * 8 + (kTh >> 4)) * 16 + (kTh & 15)] = f2bfbits(Wl[tid * 16 + i]);
        }
    }
    __syncthreads();

    const int nc = blockIdx.x & 255;
    const int b  = blockIdx.x >> 8;
    const int w    = tid >> 5;
    const int lane = tid & 31;
    const int lhi  = lane >> 4;
    const int llo  = lane & 15;

    union BF { v16bf v; uint4 q[2]; unsigned short u[16]; };
    BF Bf[8];                                  // [kk4][ct]
#pragma unroll
    for (int kk4 = 0; kk4 < 4; ++kk4)
#pragma unroll
        for (int ct = 0; ct < 2; ++ct) {
            const int n = ct * 16 + llo;
            const unsigned short* bp = ldsW + (n * 8 + kk4 * 2 + lhi) * 16;
            Bf[kk4 * 2 + ct].q[0] = *(const uint4*)(bp);
            Bf[kk4 * 2 + ct].q[1] = *(const uint4*)(bp + 8);
        }

    const float bb0 = bl[llo];
    const float bb1 = bl[16 + llo];
    const int outBase = (b * NC + nc) * Cc;

    for (int strip = 0; strip < 4; ++strip) {
        const int npBase = strip * 128 + w * 16;
        const int row    = npBase + llo;
        const float* zrow = z + (((b * NP + row) * NC + nc) * Ee);

        v8f c0 = {}; v8f c1 = {};
#pragma unroll
        for (int kk4 = 0; kk4 < 4; ++kk4) {
            const int koff = kk4 * 32 + lhi * 16;
            float t[16];
            *(float4*)(t + 0)  = *(const float4*)(zrow + koff);
            *(float4*)(t + 4)  = *(const float4*)(zrow + koff + 4);
            *(float4*)(t + 8)  = *(const float4*)(zrow + koff + 8);
            *(float4*)(t + 12) = *(const float4*)(zrow + koff + 12);
            BF A;
#pragma unroll
            for (int i = 0; i < 16; ++i) A.u[i] = f2bfbits(t[i]);
            c0 = __builtin_amdgcn_wmma_f32_16x16x32_bf16(false, A.v, false, Bf[kk4 * 2 + 0].v, (short)0, c0, false, false);
            c1 = __builtin_amdgcn_wmma_f32_16x16x32_bf16(false, A.v, false, Bf[kk4 * 2 + 1].v, (short)0, c1, false, false);
        }

        const int np0 = npBase + lhi * 8;      // 8 consecutive np -> 16B store
        union { uint4 q; unsigned short u[8]; } st;
#pragma unroll
        for (int r = 0; r < 8; ++r) st.u[r] = f2bfbits(c0[r] + bb0);
        *(uint4*)(S + ((outBase + llo) * NP + np0)) = st.q;
#pragma unroll
        for (int r = 0; r < 8; ++r) st.u[r] = f2bfbits(c1[r] + bb1);
        *(uint4*)(S + ((outBase + 16 + llo) * NP + np0)) = st.q;
    }
}

// ---------------------------------------------------------------------------
// Fused: softmax over np (row of 512), + smooth_finite distance-bucket bias
// (reshape bug => flat index f = c*131072 + nc*512 + np into natural order),
// + scrambled tile/view mask (z_mask_flat[g mod 2^19]). In-place bf16 on S.
// One wave per row, 8 rows per block.
// ---------------------------------------------------------------------------
__global__ void k_softmax(unsigned short* __restrict__ S, const float* __restrict__ dis,
                          const int* __restrict__ zmask, const float* __restrict__ Wd,
                          const float* __restrict__ bd) {
    __shared__ float ldsW[20 * 32];
    __shared__ float ldsBd[32];
    const int tid = threadIdx.x;
    if (tid < 640) ldsW[tid] = Wd[tid];
    if (tid < 32)  ldsBd[tid] = bd[tid];
    __syncthreads();

    const int row  = blockIdx.x * 8 + (tid >> 5);  // row = (b*256+nc)*32 + c
    const int lane = tid & 31;
    const int b  = row >> 13;
    const int nc = (row >> 5) & 255;
    const int c  = row & 31;
    unsigned short* rp = S + row * NP;

    float x[16];
    float mx = -3.4e38f;
#pragma unroll
    for (int j = 0; j < 16; ++j) { x[j] = bf2f(rp[j * 32 + lane]); mx = fmaxf(mx, x[j]); }
    for (int o = 16; o >= 1; o >>= 1) mx = fmaxf(mx, __shfl_xor(mx, o, 32));
    float sum = 0.0f;
#pragma unroll
    for (int j = 0; j < 16; ++j) { x[j] = __expf(x[j] - mx); sum += x[j]; }
    for (int o = 16; o >= 1; o >>= 1) sum += __shfl_xor(sum, o, 32);
    const float inv = 1.0f / sum;

    const float step  = 15.0f / 21.0f;
    const float istep = 21.0f / 15.0f;
    const float KK = 1.14136f * 7.3890560989306495f;   // 1.14136 * e^2

    for (int j = 0; j < 16; ++j) {
        const int np = j * 32 + lane;
        const int f  = c * 131072 + nc * 512 + np;     // flat idx, [NP,NC,C] natural
        const int np0 = f >> 13;                       // / (NC*C)
        const int nc0 = (f >> 5) & 255;                // c0 = f & 31 == lane
        const float d = dis[(b * NP + np0) * NC + nc0];
        float acc = ldsBd[lane];
#pragma unroll
        for (int t = 0; t < 20; ++t) {
            const float center = (float)(t + 1) * step;
            const float diff = (d - center) * istep;
            const float y1 = diff + 1.0f, y2 = 1.0f - diff;
            const float s1 = (y1 > 0.0f) ? __expf(-1.0f / y1) : 0.0f;
            const float s2 = (y2 > 0.0f) ? __expf(-1.0f / y2) : 0.0f;
            acc += KK * s1 * s2 * ldsW[t * 32 + lane];
        }
        const unsigned g = ((unsigned)((b * Cc + c) * NC + nc)) * NP + np;
        const int m = zmask[g & 0x7FFFFu];             // B*NP*NC = 2^19
        const float v = m ? 1e-9f : (x[j] * inv + acc);
        rp[j * 32 + lane] = f2bfbits(v);
    }
}

// ---------------------------------------------------------------------------
// Pair-energy heads: per (b,np,nc) dot over C=32 for gate/energy (x2 heads),
// masked, summed into acc[b] / acc[4+b].
// ---------------------------------------------------------------------------
__global__ void k_pair(const unsigned short* __restrict__ S, const int* __restrict__ zmask,
                       const float* __restrict__ Wg, const float* __restrict__ bg,
                       const float* __restrict__ We, const float* __restrict__ be,
                       const float* __restrict__ Wgp, const float* __restrict__ bgp,
                       const float* __restrict__ Wep, const float* __restrict__ bep,
                       float* __restrict__ acc) {
    __shared__ float red[256];
    const int tid = threadIdx.x;
    const int b  = blockIdx.x >> 8;
    const int nc = blockIdx.x & 255;
    const unsigned short* base = S + ((b * NC + nc) * Cc) * NP;

    float ge[2] = {0, 0}, en[2] = {0, 0}, gp[2] = {0, 0}, ep[2] = {0, 0};
#pragma unroll 4
    for (int c = 0; c < Cc; ++c) {
        const float wg = Wg[c], we = We[c], wgp = Wgp[c], wep = Wep[c];
        const float v0 = bf2f(base[c * NP + tid]);
        const float v1 = bf2f(base[c * NP + tid + 256]);
        ge[0] += v0 * wg;  en[0] += v0 * we;  gp[0] += v0 * wgp;  ep[0] += v0 * wep;
        ge[1] += v1 * wg;  en[1] += v1 * we;  gp[1] += v1 * wgp;  ep[1] += v1 * wep;
    }
    const float bgv = bg[0], bev = be[0], bgpv = bgp[0], bepv = bep[0];
    float pe = 0.0f, pep = 0.0f;
#pragma unroll
    for (int h = 0; h < 2; ++h) {
        const int np = tid + h * 256;
        const float m = zmask[(b * NP + np) * NC + nc] ? 1.0f : 0.0f;
        pe  += (1.0f / (1.0f + __expf(-(ge[h] + bgv))))  * (en[h] + bev)  * m;
        pep += (1.0f / (1.0f + __expf(-(gp[h] + bgpv)))) * (ep[h] + bepv) * m;
    }
    red[tid] = pe; __syncthreads();
    for (int s = 128; s > 0; s >>= 1) { if (tid < s) red[tid] += red[tid + s]; __syncthreads(); }
    if (tid == 0) atomicAdd(&acc[b], red[0]);
    __syncthreads();
    red[tid] = pep; __syncthreads();
    for (int s = 128; s > 0; s >>= 1) { if (tid < s) red[tid] += red[tid + s]; __syncthreads(); }
    if (tid == 0) atomicAdd(&acc[4 + b], red[0]);
}

__global__ void k_final(const float* __restrict__ acc, const float* __restrict__ bias,
                        const float* __restrict__ biasp, float* __restrict__ out) {
    const int t = threadIdx.x;
    if (t < 4) {
        const float v  = bias[0]  + acc[t];
        const float v2 = biasp[0] + acc[4 + t];
        out[Bn * NC * Ee + t]     = (v  >= 0.0f) ? v  : 0.01f * v;   // leaky_relu
        out[Bn * NC * Ee + 4 + t] = (v2 >= 0.0f) ? v2 : 0.01f * v2;
    }
}

// ---------------------------------------------------------------------------
// GEMM2 (einsum bcln,bne->bcle): per b, M = NC*C = 8192 rows (row = nc*32+c,
// contiguous K=np=512 in S), N = E = 128, K = 512. Output lig[b][nc*32+c][e]
// bf16 == exactly the [nc, c*E+e] reshape the reference feeds to W_out.
// Block: 64 rows x 128 cols, 8 waves (4x2), 4 WMMA tiles per wave.
// LDS B tile pre-swizzled: frag(n, khalf) = 32 contiguous bytes.
// ---------------------------------------------------------------------------
__global__ void k_gemm2(const unsigned short* __restrict__ zc, const float* __restrict__ pr,
                        unsigned short* __restrict__ lig) {
    __shared__ unsigned short ldsB[32 * 128];   // [(n*2 + khalf)*16 + (k&15)]
    const int tid = threadIdx.x;
    const int b  = blockIdx.x >> 7;
    const int rb = (blockIdx.x & 127) * 64;
    const int w = tid >> 5, lane = tid & 31;
    const int wr = w >> 1, wc = w & 1;
    const int rowBase = rb + wr * 16;
    const int lhi = lane >> 4, llo = lane & 15;

    v8f acc[4] = {{}, {}, {}, {}};
    const unsigned short* arow = zc + (b * (NC * Cc) + rowBase + llo) * NP;

    const int kStage = (tid * 16) >> 7;   // this thread's fixed k within tile
    const int nStage = (tid * 16) & 127;  // 16 consecutive n

    for (int kk = 0; kk < NP; kk += 32) {
        __syncthreads();
        const float* src = pr + b * (NP * Ee) + kk * Ee + tid * 16;
#pragma unroll
        for (int i = 0; i < 16; ++i) {
            const int n = nStage + i;
            ldsB[((n << 1) + (kStage >> 4)) * 16 + (kStage & 15)] = f2bfbits(src[i]);
        }
        __syncthreads();

        const int koff = kk + lhi * 16;
        union { v16bf v; uint4 q[2]; } A;
        A.q[0] = *(const uint4*)(arow + koff);
        A.q[1] = *(const uint4*)(arow + koff + 8);

#pragma unroll
        for (int t = 0; t < 4; ++t) {
            const int n = wc * 64 + t * 16 + llo;
            const unsigned short* bp = ldsB + ((n << 1) + lhi) * 16;
            union { v16bf v; uint4 q[2]; } Bf;
            Bf.q[0] = *(const uint4*)(bp);
            Bf.q[1] = *(const uint4*)(bp + 8);
            acc[t] = __builtin_amdgcn_wmma_f32_16x16x32_bf16(false, A.v, false, Bf.v, (short)0, acc[t], false, false);
        }
    }
#pragma unroll
    for (int t = 0; t < 4; ++t) {
        const int n = wc * 64 + t * 16 + llo;
#pragma unroll
        for (int r = 0; r < 8; ++r) {
            const int ro = rowBase + lhi * 8 + r;
            lig[(b * (NC * Cc) + ro) * Ee + n] = f2bfbits(acc[t][r]);
        }
    }
}

// ---------------------------------------------------------------------------
// GEMM3: ligand_rep[row=b*256+nc][e] = lig[row][0:4096] @ W_out + b_out
// M = 1024, K = 4096, N = 128. Same structure / swizzled LDS as GEMM2.
// ---------------------------------------------------------------------------
__global__ void k_gemm3(const unsigned short* __restrict__ lig, const float* __restrict__ Wo,
                        const float* __restrict__ bo, float* __restrict__ out) {
    __shared__ unsigned short ldsB[32 * 128];
    const int tid = threadIdx.x;
    const int rb = blockIdx.x * 64;
    const int w = tid >> 5, lane = tid & 31;
    const int wr = w >> 1, wc = w & 1;
    const int rowBase = rb + wr * 16;
    const int lhi = lane >> 4, llo = lane & 15;

    v8f acc[4] = {{}, {}, {}, {}};
    const unsigned short* arow = lig + (rowBase + llo) * (Cc * Ee);

    const int kStage = (tid * 16) >> 7;
    const int nStage = (tid * 16) & 127;

    for (int kk = 0; kk < Cc * Ee; kk += 32) {
        __syncthreads();
        const float* src = Wo + kk * Ee + tid * 16;
#pragma unroll
        for (int i = 0; i < 16; ++i) {
            const int n = nStage + i;
            ldsB[((n << 1) + (kStage >> 4)) * 16 + (kStage & 15)] = f2bfbits(src[i]);
        }
        __syncthreads();

        const int koff = kk + lhi * 16;
        __builtin_prefetch(arow + koff + 256, 0, 1);   // global_prefetch_b8 on A stream
        union { v16bf v; uint4 q[2]; } A;
        A.q[0] = *(const uint4*)(arow + koff);
        A.q[1] = *(const uint4*)(arow + koff + 8);

#pragma unroll
        for (int t = 0; t < 4; ++t) {
            const int n = wc * 64 + t * 16 + llo;
            const unsigned short* bp = ldsB + ((n << 1) + lhi) * 16;
            union { v16bf v; uint4 q[2]; } Bf;
            Bf.q[0] = *(const uint4*)(bp);
            Bf.q[1] = *(const uint4*)(bp + 8);
            acc[t] = __builtin_amdgcn_wmma_f32_16x16x32_bf16(false, A.v, false, Bf.v, (short)0, acc[t], false, false);
        }
    }
#pragma unroll
    for (int t = 0; t < 4; ++t) {
        const int n = wc * 64 + t * 16 + llo;
        const float bb = bo[n];
#pragma unroll
        for (int r = 0; r < 8; ++r) {
            const int ro = rowBase + lhi * 8 + r;
            out[ro * Ee + n] = acc[t][r] + bb;
        }
    }
}

// ---------------------------------------------------------------------------
extern "C" void kernel_launch(void* const* d_in, const int* in_sizes, int n_in,
                              void* d_out, int out_size, void* d_ws, size_t ws_size,
                              hipStream_t stream) {
    (void)in_sizes; (void)n_in; (void)out_size; (void)ws_size;
    const float* z     = (const float*)d_in[0];
    const int*   zmask = (const int*)  d_in[1];
    const float* pr    = (const float*)d_in[2];
    const float* dis   = (const float*)d_in[3];
    const float* Wl    = (const float*)d_in[4];
    const float* bl    = (const float*)d_in[5];
    const float* Wd    = (const float*)d_in[6];
    const float* bd    = (const float*)d_in[7];
    const float* We    = (const float*)d_in[8];
    const float* be    = (const float*)d_in[9];
    const float* Wg    = (const float*)d_in[10];
    const float* bg    = (const float*)d_in[11];
    const float* bias  = (const float*)d_in[12];
    const float* Wep   = (const float*)d_in[13];
    const float* bep   = (const float*)d_in[14];
    const float* Wgp   = (const float*)d_in[15];
    const float* bgp   = (const float*)d_in[16];
    const float* biasp = (const float*)d_in[17];
    const float* Wo    = (const float*)d_in[18];
    const float* bo    = (const float*)d_in[19];
    float* out = (float*)d_out;

    char* ws = (char*)d_ws;
    unsigned short* S   = (unsigned short*)ws;                       // 33,554,432 B (zc bf16)
    unsigned short* lig = (unsigned short*)(ws + 33554432);          //  8,388,608 B
    float*          acc = (float*)(ws + 33554432 + 8388608);         //        32 B

    k_zero   <<<1,    32,  0, stream>>>(acc);
    k_gemm1  <<<1024, 256, 0, stream>>>(z, Wl, bl, S);
    k_softmax<<<4096, 256, 0, stream>>>(S, dis, zmask, Wd, bd);
    k_pair   <<<1024, 256, 0, stream>>>(S, zmask, Wg, bg, We, be, Wgp, bgp, Wep, bep, acc);
    k_final  <<<1,    32,  0, stream>>>(acc, bias, biasp, out);
    k_gemm2  <<<512,  256, 0, stream>>>(S, pr, lig);
    k_gemm3  <<<16,   256, 0, stream>>>(lig, Wo, bo, out);
}
